// SuperPoint_19756849562297
// MI455X (gfx1250) — matrix operand order repaired and checked by reading, not compiled, not run on hardware
//
#include <hip/hip_runtime.h>

typedef __bf16 bf16_t;
typedef bf16_t v16bf __attribute__((ext_vector_type(16)));
typedef float  v8f   __attribute__((ext_vector_type(8)));
typedef unsigned short u16;
typedef unsigned int   u32;
typedef u32 v4u __attribute__((ext_vector_type(4)));

union Frag {
  v4u   q[2];
  v16bf v;
};

// ---------- bf16 helpers ----------
static __device__ __forceinline__ u16 f2bf(float f) {
  unsigned int u = __float_as_uint(f);
  unsigned int r = u + 0x7FFFu + ((u >> 16) & 1u);
  return (u16)(r >> 16);
}
static __device__ __forceinline__ float bf2f(u16 h) {
  return __uint_as_float(((unsigned int)h) << 16);
}

// ---------- weight repack: OIHW fp32 -> [Opad][Ktot] bf16, K = kpos*C + c ----------
__global__ void cvtw_kernel(const float* __restrict__ w, u16* __restrict__ q,
                            int C, int O, int Opad, int ks, int Ktot, int total) {
  int idx = blockIdx.x * blockDim.x + threadIdx.x;
  if (idx >= total) return;
  int k = idx % Ktot;
  int o = idx / Ktot;
  int c  = k % C;
  int kp = k / C;
  int ky = kp / ks, kx = kp % ks;
  float v = 0.0f;
  if (o < O) v = w[(((size_t)o * C + c) * ks + ky) * ks + kx];
  q[idx] = f2bf(v);
}

// ---------- conv1a: 1->64 channels, scalar (tiny fraction of FLOPs) ----------
__global__ void conv1a_kernel(const float* __restrict__ img,
                              const float* __restrict__ w,
                              const float* __restrict__ bias,
                              u16* __restrict__ out) {
  size_t idx = (size_t)blockIdx.x * blockDim.x + threadIdx.x;
  const size_t total = (size_t)8 * 512 * 640 * 64;
  if (idx >= total) return;
  int o = (int)(idx & 63);
  size_t pix = idx >> 6;
  int x = (int)(pix % 640);
  int y = (int)((pix / 640) % 512);
  int b = (int)(pix / (640 * 512));
  float s = bias[o];
  #pragma unroll
  for (int ky = 0; ky < 3; ++ky)
    #pragma unroll
    for (int kx = 0; kx < 3; ++kx) {
      int gy = y + ky - 1, gx = x + kx - 1;
      if (gy >= 0 && gy < 512 && gx >= 0 && gx < 640)
        s += img[((size_t)b * 512 + gy) * 640 + gx] * w[o * 9 + ky * 3 + kx];
    }
  s = fmaxf(s, 0.0f);
  out[pix * 64 + o] = f2bf(s);
}

// ---------- generic implicit-GEMM conv via WMMA bf16 ----------
// in : NHWC bf16 (channel stride C = 1<<CSHIFT)
// wq : [Opad][Ktot] bf16, Ktot = KS*KS*C (K contiguous per output channel)
// Each wave: M = 32 pixels (two 16x16 accumulators sharing one B fragment)
//            N = 16 output channels.  4 waves/block = 64-channel tile group.
// A fragment = 2x ds_read_b128, B fragment = 2x global_load_b128, 2 WMMA/chunk.
template <int KS, int CSHIFT>
__global__ __launch_bounds__(128)
void conv_wmma(const u16* __restrict__ in, const u16* __restrict__ wq,
               const float* __restrict__ bias, void* __restrict__ outp,
               int H, int W, int O, int Opad,
               int relu, int out_fp32, int tgShift, int tgMask) {
  constexpr int C    = 1 << CSHIFT;
  constexpr int PAD  = KS >> 1;
  constexpr int LDW  = 32 + 2 * PAD;          // 32 or 34
  constexpr int LDSN = KS * LDW * C;          // max 3*34*128 = 13056 u16
  constexpr int KTOT = KS * KS * C;
  __shared__ u16 lds[LDSN];

  const int tid  = threadIdx.x;
  const int lane = tid & 31;
  const int wave = tid >> 5;
  const int x0 = blockIdx.x * 32;
  const int y  = blockIdx.y;
  const int b  = blockIdx.z >> tgShift;
  const int tg = blockIdx.z & tgMask;

  // Stage input patch (KS rows x LDW cols x C) into LDS, zero-padded.
  for (int i = tid; i < LDSN; i += 128) {
    int c  = i & (C - 1);
    int t  = i >> CSHIFT;
    int xx = t % LDW;
    int dy = t / LDW;
    int gy = y + dy - PAD;
    int gx = x0 + xx - PAD;
    u16 v = 0;
    if (gy >= 0 && gy < H && gx >= 0 && gx < W)
      v = in[(((size_t)b * H + gy) * (size_t)W + gx) * C + c];
    lds[i] = v;
  }
  __syncthreads();

  const int nt = tg * 4 + wave;         // wave-uniform
  if (nt * 16 >= Opad) return;          // whole wave exits: EXEC stays all-1s

  const int half = lane >> 4;
  const int mn   = lane & 15;
  const int o    = nt * 16 + mn;        // output channel (also weight row)

  v8f acc0, acc1;
  #pragma unroll
  for (int i = 0; i < 8; ++i) { acc0[i] = 0.0f; acc1[i] = 0.0f; }

  // B stream: for fixed n = mn, lane needs K = kglobal + half*8 + [0..7] and
  // kglobal + 16 + half*8 + [0..7] in the [Opad][Ktot] layout; advances by 32 per chunk.
  const u16* bptr = wq + (size_t)o * KTOT + half * 8;

  #pragma unroll
  for (int kpos = 0; kpos < KS * KS; ++kpos) {
    constexpr int CH = C >> 5;          // chunks per tap: 2, 4 or 8
    const int dy = kpos / KS, dx = kpos % KS;
    // A stream: lane holds M = mn (tile0) / 16+mn (tile1); within a 32-wide
    // chunk the lane's elements are k = half*8+[0..7] and 16+half*8+[0..7].
    const u16* aptr0 = &lds[((dy * LDW + mn + dx) << CSHIFT)] + half * 8;
    const u16* aptr1 = aptr0 + (16 << CSHIFT);
    __builtin_prefetch((const void*)(bptr + C), 0, 1);  // next tap's weights
    #pragma unroll
    for (int cb = 0; cb < CH; ++cb) {
      Frag A0, A1, Bf;
      A0.q[0] = *(const v4u*)(aptr0);
      A0.q[1] = *(const v4u*)(aptr0 + 16);
      A1.q[0] = *(const v4u*)(aptr1);
      A1.q[1] = *(const v4u*)(aptr1 + 16);
      Bf.q[0] = *(const v4u*)(bptr);
      Bf.q[1] = *(const v4u*)(bptr + 16);
      acc0 = __builtin_amdgcn_wmma_f32_16x16x32_bf16(false, A0.v, false, Bf.v,
                                                     (short)0, acc0, false, false);
      acc1 = __builtin_amdgcn_wmma_f32_16x16x32_bf16(false, A1.v, false, Bf.v,
                                                     (short)0, acc1, false, false);
      aptr0 += 32;
      aptr1 += 32;
      bptr  += 32;
    }
  }

  // C/D layout: n = lane&15, m = r + half*8 (+16 for tile1)
  if (o >= O) return;
  const float bv = bias[o];
  #pragma unroll
  for (int r = 0; r < 8; ++r) {
    const int m = r + half * 8;
    {
      const int x = x0 + m;
      if (x < W) {
        float val = acc0[r] + bv;
        if (relu) val = fmaxf(val, 0.0f);
        size_t pix = ((size_t)b * H + y) * (size_t)W + x;
        if (out_fp32) ((float*)outp)[pix * O + o] = val;
        else          ((u16*)outp)[pix * O + o]   = f2bf(val);
      }
    }
    {
      const int x = x0 + 16 + m;
      if (x < W) {
        float val = acc1[r] + bv;
        if (relu) val = fmaxf(val, 0.0f);
        size_t pix = ((size_t)b * H + y) * (size_t)W + x;
        if (out_fp32) ((float*)outp)[pix * O + o] = val;
        else          ((u16*)outp)[pix * O + o]   = f2bf(val);
      }
    }
  }
}

// ---------- 2x2 max pool, NHWC bf16 ----------
__global__ void pool2_kernel(const u16* __restrict__ in, u16* __restrict__ out,
                             int Ho, int Wo, int C) {
  size_t idx = (size_t)blockIdx.x * blockDim.x + threadIdx.x;
  size_t total = (size_t)8 * Ho * Wo * C;
  if (idx >= total) return;
  int c = (int)(idx % C);
  size_t t = idx / C;
  int x = (int)(t % Wo); t /= Wo;
  int y = (int)(t % Ho);
  int b = (int)(t / Ho);
  int Wi = Wo * 2, Hi = Ho * 2;
  const u16* p = in + (((size_t)b * Hi + 2 * y) * Wi + 2 * x) * C + c;
  float m = bf2f(p[0]);
  m = fmaxf(m, bf2f(p[C]));
  m = fmaxf(m, bf2f(p[(size_t)Wi * C]));
  m = fmaxf(m, bf2f(p[(size_t)Wi * C + C]));
  out[idx] = f2bf(m);
}

// ---------- softmax over 65ch + drop dustbin + pixel shuffle to (B,512,640) ----------
__global__ void softmax_kernel(const float* __restrict__ sb, float* __restrict__ S) {
  int idx = blockIdx.x * blockDim.x + threadIdx.x;  // (b,hc,wc), wc fastest
  if (idx >= 8 * 64 * 80) return;
  int wc = idx % 80;
  int hc = (idx / 80) % 64;
  int b  = idx / (80 * 64);
  const float* p = sb + (size_t)idx * 65;
  float mx = p[0];
  for (int c = 1; c < 65; ++c) mx = fmaxf(mx, p[c]);
  float sum = 0.0f;
  float e[65];
  for (int c = 0; c < 65; ++c) { e[c] = expf(p[c] - mx); sum += e[c]; }
  float inv = 1.0f / sum;
  for (int c = 0; c < 64; ++c) {
    int i = c >> 3, j = c & 7;
    S[((size_t)b * 512 + hc * 8 + i) * 640 + wc * 8 + j] = e[c] * inv;
  }
}

// ---------- separable 9x9 dilation (SAME, -inf pad == skip OOB) ----------
__global__ void rowmax9_kernel(const float* __restrict__ in, float* __restrict__ out) {
  size_t idx = (size_t)blockIdx.x * blockDim.x + threadIdx.x;
  if (idx >= (size_t)8 * 512 * 640) return;
  int x = (int)(idx % 640);
  size_t base = idx - x;
  float m = -3.0e38f;
  for (int d = -4; d <= 4; ++d) {
    int xx = x + d;
    if (xx >= 0 && xx < 640) m = fmaxf(m, in[base + xx]);
  }
  out[idx] = m;
}
__global__ void colmax9_kernel(const float* __restrict__ in, float* __restrict__ out) {
  size_t idx = (size_t)blockIdx.x * blockDim.x + threadIdx.x;
  if (idx >= (size_t)8 * 512 * 640) return;
  int x = (int)(idx % 640);
  int y = (int)((idx / 640) % 512);
  int b = (int)(idx / (640 * 512));
  float m = -3.0e38f;
  for (int d = -4; d <= 4; ++d) {
    int yy = y + d;
    if (yy >= 0 && yy < 512) m = fmaxf(m, in[((size_t)b * 512 + yy) * 640 + x]);
  }
  out[idx] = m;
}

// ---------- NMS elementwise steps ----------
__global__ void eq_kernel(const float* __restrict__ a, const float* __restrict__ b,
                          float* __restrict__ m) {
  size_t idx = (size_t)blockIdx.x * blockDim.x + threadIdx.x;
  if (idx >= (size_t)8 * 512 * 640) return;
  m[idx] = (a[idx] == b[idx]) ? 1.0f : 0.0f;
}
__global__ void supp_kernel(const float* __restrict__ mpmask, const float* __restrict__ S,
                            float* __restrict__ supps) {
  size_t idx = (size_t)blockIdx.x * blockDim.x + threadIdx.x;
  if (idx >= (size_t)8 * 512 * 640) return;
  supps[idx] = (mpmask[idx] > 0.0f) ? 0.0f : S[idx];
}
__global__ void upd_kernel(float* __restrict__ mask, const float* __restrict__ mpmask,
                           const float* __restrict__ supps, const float* __restrict__ mps) {
  size_t idx = (size_t)blockIdx.x * blockDim.x + threadIdx.x;
  if (idx >= (size_t)8 * 512 * 640) return;
  bool supp  = mpmask[idx] > 0.0f;
  bool nmax  = supps[idx] == mps[idx];
  if (mask[idx] == 0.0f && nmax && !supp) mask[idx] = 1.0f;
}
__global__ void sel_kernel(const float* __restrict__ mask, const float* __restrict__ S,
                           float* __restrict__ sel) {
  size_t idx = (size_t)blockIdx.x * blockDim.x + threadIdx.x;
  if (idx >= (size_t)8 * 512 * 640) return;
  int x = (int)(idx % 640);
  int y = (int)((idx / 640) % 512);
  float v = (mask[idx] != 0.0f) ? S[idx] : 0.0f;
  bool keep = (v > 0.005f) && (x >= 4) && (x < 636) && (y >= 4) && (y < 508);
  sel[idx] = keep ? v : 0.0f;
}

// ---------- exact iterative top-1024 per image ----------
__global__ __launch_bounds__(1024)
void topk_kernel(float* __restrict__ sel, float* __restrict__ out_kp,
                 float* __restrict__ out_sc) {
  const int b = blockIdx.x;
  const int tid = threadIdx.x;
  const int N = 512 * 640;
  const int SEG = N / 1024;  // 320
  float* s = sel + (size_t)b * N;
  __shared__ float sval[1024];
  __shared__ int   sidx[1024];
  __shared__ float rval[1024];
  __shared__ int   ridx[1024];
  __shared__ int   s_w;

  {
    float bv = -3.0e38f; int bi = tid * SEG;
    for (int i = tid * SEG; i < tid * SEG + SEG; ++i) {
      float v = s[i];
      if (v > bv) { bv = v; bi = i; }   // ascending scan keeps lowest index on tie
    }
    sval[tid] = bv; sidx[tid] = bi;
  }
  __syncthreads();

  for (int k = 0; k < 1024; ++k) {
    rval[tid] = sval[tid]; ridx[tid] = sidx[tid];
    __syncthreads();
    for (int off = 512; off > 0; off >>= 1) {
      if (tid < off) {
        float v2 = rval[tid + off]; int i2 = ridx[tid + off];
        float v1 = rval[tid];       int i1 = ridx[tid];
        if (v2 > v1 || (v2 == v1 && i2 < i1)) { rval[tid] = v2; ridx[tid] = i2; }
      }
      __syncthreads();
    }
    if (tid == 0) {
      int wi = ridx[0]; float wv = rval[0];
      out_kp[((size_t)b * 1024 + k) * 2 + 0] = (float)(wi % 640);
      out_kp[((size_t)b * 1024 + k) * 2 + 1] = (float)(wi / 640);
      out_sc[(size_t)b * 1024 + k] = wv;
      s_w = wi;
    }
    __syncthreads();
    int wi = s_w;
    if (tid == wi / SEG) {
      s[wi] = -3.0e38f;
      float bv = -3.0e38f; int bi = tid * SEG;
      for (int i = tid * SEG; i < tid * SEG + SEG; ++i) {
        float v = s[i];
        if (v > bv) { bv = v; bi = i; }
      }
      sval[tid] = bv; sidx[tid] = bi;
    }
    __syncthreads();
  }
}

// ---------- per-pixel channel L2 normalization of descriptor map (in place) ----------
__global__ __launch_bounds__(256)
void l2norm_kernel(float* __restrict__ d) {
  int pix = blockIdx.x;
  int c   = threadIdx.x;
  float v = d[(size_t)pix * 256 + c];
  __shared__ float red[256];
  __shared__ float snorm;
  red[c] = v * v;
  __syncthreads();
  for (int off = 128; off > 0; off >>= 1) {
    if (c < off) red[c] += red[c + off];
    __syncthreads();
  }
  if (c == 0) snorm = fmaxf(sqrtf(red[0]), 1e-12f);
  __syncthreads();
  d[(size_t)pix * 256 + c] = v / snorm;
}

// ---------- bilinear descriptor sampling + final L2 norm ----------
__global__ __launch_bounds__(256)
void sample_kernel(const float* __restrict__ dnorm, const float* __restrict__ kp,
                   float* __restrict__ desc_out) {
  const int k = blockIdx.x;   // keypoint
  const int b = blockIdx.y;   // image
  const int c = threadIdx.x;  // channel
  const int w = 80, h = 64;
  float kx = kp[((size_t)b * 1024 + k) * 2 + 0];
  float ky = kp[((size_t)b * 1024 + k) * 2 + 1];
  // align_corners grid math from reference (s=8):
  float ix = (kx - 3.5f) / 635.5f * (float)(w - 1);
  float iy = (ky - 3.5f) / 507.5f * (float)(h - 1);
  float x0f = floorf(ix), y0f = floorf(iy);
  float wx1 = ix - x0f, wx0 = 1.0f - wx1;
  float wy1 = iy - y0f, wy0 = 1.0f - wy1;
  float acc = 0.0f;
  #pragma unroll
  for (int q = 0; q < 4; ++q) {
    float xf = x0f + (float)(q & 1);
    float yf = y0f + (float)(q >> 1);
    float wgt = ((q & 1) ? wx1 : wx0) * ((q >> 1) ? wy1 : wy0);
    if (xf >= 0.0f && xf <= (float)(w - 1) && yf >= 0.0f && yf <= (float)(h - 1)) {
      int xc = (int)xf, yc = (int)yf;
      acc += dnorm[(((size_t)b * h + yc) * w + xc) * 256 + c] * wgt;
    }
  }
  __shared__ float red[256];
  __shared__ float snorm;
  red[c] = acc * acc;
  __syncthreads();
  for (int off = 128; off > 0; off >>= 1) {
    if (c < off) red[c] += red[c + off];
    __syncthreads();
  }
  if (c == 0) snorm = fmaxf(sqrtf(red[0]), 1e-12f);
  __syncthreads();
  desc_out[((size_t)b * 256 + c) * 1024 + k] = acc / snorm;
}

// =================== host orchestration ===================
extern "C" void kernel_launch(void* const* d_in, const int* in_sizes, int n_in,
                              void* d_out, int out_size, void* d_ws, size_t ws_size,
                              hipStream_t stream) {
  (void)in_sizes; (void)n_in; (void)out_size; (void)ws_size;
  const float* img = (const float*)d_in[0];
  auto Wp = [&](int i) { return (const float*)d_in[1 + 2 * i]; };
  auto Bp = [&](int i) { return (const float*)d_in[2 + 2 * i]; };

  char* ws = (char*)d_ws;
  const size_t MB = 1024 * 1024;
  u16*   WQ   = (u16*)(ws + 0);           // repacked weights (2.6 MB)
  u16*   bufA = (u16*)(ws + 8 * MB);      // activation ping (<=335.6 MB)
  u16*   bufB = (u16*)(ws + 344 * MB);    // activation pong
  float* SB   = (float*)(ws + 680 * MB);  // score head logits (B,64,80,65) f32
  float* DB   = (float*)(ws + 692 * MB);  // desc head output  (B,64,80,256) f32
  float* F0   = (float*)(ws + 740 * MB);  // full-res score map S
  float* F1   = (float*)(ws + 752 * MB);  // scratch (rowmax)
  float* F2   = (float*)(ws + 764 * MB);  // scratch (colmax / mp)
  float* F3   = (float*)(ws + 776 * MB);  // max_mask
  float* F4   = (float*)(ws + 788 * MB);  // supp_scores
  float* F5   = (float*)(ws + 800 * MB);  // mps scratch, then selected scores

  struct LayerDef { int di, C, cShift, O, Opad, ks, H, W, relu, ofp32; };
  const LayerDef L[11] = {
    { 1,  64, 6,  64,  64, 3, 512, 640, 1, 0},  // conv1b
    { 2,  64, 6,  64,  64, 3, 256, 320, 1, 0},  // conv2a
    { 3,  64, 6,  64,  64, 3, 256, 320, 1, 0},  // conv2b
    { 4,  64, 6, 128, 128, 3, 128, 160, 1, 0},  // conv3a
    { 5, 128, 7, 128, 128, 3, 128, 160, 1, 0},  // conv3b
    { 6, 128, 7, 128, 128, 3,  64,  80, 1, 0},  // conv4a
    { 7, 128, 7, 128, 128, 3,  64,  80, 1, 0},  // conv4b
    { 8, 128, 7, 256, 256, 3,  64,  80, 1, 0},  // convSa
    { 9, 256, 8,  65,  80, 1,  64,  80, 0, 1},  // convSb
    {10, 128, 7, 256, 256, 3,  64,  80, 1, 0},  // convDa
    {11, 256, 8, 256, 256, 1,  64,  80, 0, 1},  // convDb
  };
  size_t qoff[11]; size_t accq = 0;
  for (int i = 0; i < 11; ++i) {
    qoff[i] = accq;
    accq += (size_t)L[i].ks * L[i].ks * L[i].C * L[i].Opad;
  }
  for (int i = 0; i < 11; ++i) {
    int Ktot  = L[i].ks * L[i].ks * L[i].C;
    int total = L[i].Opad * Ktot;
    cvtw_kernel<<<(total + 255) / 256, 256, 0, stream>>>(
        Wp(L[i].di), WQ + qoff[i], L[i].C, L[i].O, L[i].Opad, L[i].ks, Ktot, total);
  }

  auto conv = [&](int li, const u16* in, void* out) {
    const LayerDef& l = L[li];
    int tg = (l.Opad + 63) / 64;            // 1, 2 or 4 (power of two)
    int tgShift = (tg == 1) ? 0 : (tg == 2) ? 1 : 2;
    dim3 grid((l.W + 31) / 32, l.H, 8 * tg);
    if (l.ks == 3 && l.cShift == 6) {
      conv_wmma<3, 6><<<grid, 128, 0, stream>>>(in, WQ + qoff[li], Bp(l.di), out,
                                                l.H, l.W, l.O, l.Opad,
                                                l.relu, l.ofp32, tgShift, tg - 1);
    } else if (l.ks == 3 && l.cShift == 7) {
      conv_wmma<3, 7><<<grid, 128, 0, stream>>>(in, WQ + qoff[li], Bp(l.di), out,
                                                l.H, l.W, l.O, l.Opad,
                                                l.relu, l.ofp32, tgShift, tg - 1);
    } else {
      conv_wmma<1, 8><<<grid, 128, 0, stream>>>(in, WQ + qoff[li], Bp(l.di), out,
                                                l.H, l.W, l.O, l.Opad,
                                                l.relu, l.ofp32, tgShift, tg - 1);
    }
  };
  auto pool = [&](const u16* in, u16* out, int Ho, int Wo, int C) {
    size_t total = (size_t)8 * Ho * Wo * C;
    pool2_kernel<<<(unsigned)((total + 255) / 256), 256, 0, stream>>>(in, out, Ho, Wo, C);
  };

  // ---- backbone ----
  {
    size_t total = (size_t)8 * 512 * 640 * 64;
    conv1a_kernel<<<(unsigned)((total + 255) / 256), 256, 0, stream>>>(
        img, Wp(0), Bp(0), bufA);
  }
  conv(0, bufA, bufB);             // conv1b
  pool(bufB, bufA, 256, 320, 64);
  conv(1, bufA, bufB);             // conv2a
  conv(2, bufB, bufA);             // conv2b
  pool(bufA, bufB, 128, 160, 64);
  conv(3, bufB, bufA);             // conv3a
  conv(4, bufA, bufB);             // conv3b
  pool(bufB, bufA, 64, 80, 128);
  conv(5, bufA, bufB);             // conv4a
  conv(6, bufB, bufA);             // conv4b -> enc in bufA

  // ---- score head ----
  conv(7, bufA, bufB);             // convSa (bf16)
  conv(8, bufB, SB);               // convSb -> fp32 logits (stride 65)

  const size_t NTOT = (size_t)8 * 512 * 640;
  const unsigned EB = (unsigned)((NTOT + 255) / 256);
  softmax_kernel<<<(8 * 64 * 80 + 255) / 256, 256, 0, stream>>>(SB, F0);

  // simple_nms: initial mask
  rowmax9_kernel<<<EB, 256, 0, stream>>>(F0, F1);
  colmax9_kernel<<<EB, 256, 0, stream>>>(F1, F2);
  eq_kernel<<<EB, 256, 0, stream>>>(F0, F2, F3);
  // two suppression iterations
  for (int it = 0; it < 2; ++it) {
    rowmax9_kernel<<<EB, 256, 0, stream>>>(F3, F1);
    colmax9_kernel<<<EB, 256, 0, stream>>>(F1, F2);      // mp(mask)
    supp_kernel<<<EB, 256, 0, stream>>>(F2, F0, F4);     // supp_scores
    rowmax9_kernel<<<EB, 256, 0, stream>>>(F4, F1);
    colmax9_kernel<<<EB, 256, 0, stream>>>(F1, F5);      // mp(supp_scores)
    upd_kernel<<<EB, 256, 0, stream>>>(F3, F2, F4, F5);  // mask |= newmax & ~supp
  }
  sel_kernel<<<EB, 256, 0, stream>>>(F3, F0, F5);        // threshold + border

  float* kp_out   = (float*)d_out;                 // (8,1024,2)
  float* sc_out   = kp_out + (size_t)8 * 1024 * 2; // (8,1024)
  float* desc_out = sc_out + (size_t)8 * 1024;     // (8,256,1024)
  topk_kernel<<<8, 1024, 0, stream>>>(F5, kp_out, sc_out);

  // ---- descriptor head ----
  conv(9, bufA, bufB);             // convDa (bf16)
  conv(10, bufB, DB);              // convDb -> fp32 (stride 256)
  l2norm_kernel<<<8 * 64 * 80, 256, 0, stream>>>(DB);
  {
    dim3 grid(1024, 8);
    sample_kernel<<<grid, 256, 0, stream>>>(DB, kp_out, desc_out);
  }
}
